// SelfAttention_52493090292139
// MI455X (gfx1250) — matrix operand order repaired and checked by reading, not compiled, not run on hardware
//
#include <hip/hip_runtime.h>

// ---------------------------------------------------------------------------
// CDNA5 (gfx1250) GQA attention layer.
// Matmuls: v_wmma_f32_16x16x32_bf16.  GEMM tiles fed by the Tensor Data Mover
// (tensor_load_to_lds + s_wait_tensorcnt) with LDS row padding, double-buffered.
// ---------------------------------------------------------------------------

typedef __attribute__((ext_vector_type(16))) __bf16 v16bf;
typedef __attribute__((ext_vector_type(8)))  __bf16 v8bf;
typedef __attribute__((ext_vector_type(8)))  float  v8f;

#define DIM_   4096
#define SEQ_   2048
#define BATCH_ 2
#define NH_    32
#define NKV_   8
#define HD_    128
#define ROWS_  (BATCH_ * SEQ_)   // 4096

static __device__ __forceinline__ v16bf cat8(v8bf lo, v8bf hi) {
    return __builtin_shufflevector(lo, hi, 0,1,2,3,4,5,6,7,8,9,10,11,12,13,14,15);
}
static __device__ __forceinline__ v8f wmma_bf16(v16bf a, v16bf b, v8f c) {
    // (neg_a, A, neg_b, B, c_mod, C, reuse_a, reuse_b)
    return __builtin_amdgcn_wmma_f32_16x16x32_bf16(false, a, false, b, (short)0, c,
                                                   false, false);
}

// ----------------------------- TDM tile loader -----------------------------
#if __has_builtin(__builtin_amdgcn_tensor_load_to_lds)
#define HAVE_TDM 1
typedef __attribute__((ext_vector_type(4))) unsigned int u32x4;
typedef __attribute__((ext_vector_type(4))) int          i32x4;
typedef __attribute__((ext_vector_type(8))) int          i32x8;

// Load a (tile_rows x 32)-element bf16 tile (row stride = rowlen elements) from
// global to LDS, padding each 64B LDS row with 16B -> 80B stride (40 halves).
static __device__ __forceinline__ void tdm_tile_to_lds(
        const __bf16* gsrc, unsigned lds_off, int rowlen, int nrows, int tile_rows) {
    unsigned long long ga = (unsigned long long)(size_t)gsrc;
    u32x4 g0;
    g0.x = 1u;                                       // count=1, user descriptor
    g0.y = lds_off;                                  // lds_addr (bytes)
    g0.z = (unsigned)ga;                             // global_addr[31:0]
    g0.w = (unsigned)((ga >> 32) & 0x1FFFFFFull)     // global_addr[56:32]
         | (2u << 30);                               // type = 2 ("image")
    i32x8 g1;
    g1[0] = (int)((1u << 16)      // data_size = 1 -> 2 bytes
                | (1u << 20)      // pad_enable
                | (3u << 22)      // pad_interval: 16 DWORDs (64B) per row
                | (3u << 25));    // pad_amount:   4 DWORDs (16B)
    g1[1] = (int)(((unsigned)rowlen & 0xFFFFu) << 16);            // tensor_dim0 lo
    g1[2] = (int)((((unsigned)rowlen >> 16) & 0xFFFFu)            // tensor_dim0 hi
                | (((unsigned)nrows & 0xFFFFu) << 16));           // tensor_dim1 lo
    g1[3] = (int)((((unsigned)nrows >> 16) & 0xFFFFu)             // tensor_dim1 hi
                | (32u << 16));                                   // tile_dim0 = 32
    g1[4] = tile_rows;                                            // tile_dim1
    g1[5] = rowlen;                                               // dim0_stride lo
    g1[6] = 0;
    g1[7] = 0;
    i32x4 z4 = {0, 0, 0, 0};
#if __has_include(<hip/amd_detail/amd_gfx1250_TDM.h>)
    i32x8 z8 = {0, 0, 0, 0, 0, 0, 0, 0};
    __builtin_amdgcn_tensor_load_to_lds(g0, g1, z4, z4, z8, 0);   // 6-arg toolchain
#else
    __builtin_amdgcn_tensor_load_to_lds(g0, g1, z4, z4, 0);       // ROCm 7.2 5-arg
#endif
}
#endif  // HAVE_TDM

// ------------------------------ f32 -> bf16 --------------------------------
__global__ void cast_f32_bf16(const float* __restrict__ in,
                              __bf16* __restrict__ out, int n) {
    for (int i = blockIdx.x * blockDim.x + threadIdx.x; i < n;
         i += gridDim.x * blockDim.x)
        out[i] = (__bf16)in[i];
}

// ------------------- transpose + cast: out[C][R] = in[R][C] ----------------
__global__ __launch_bounds__(256) void transpose_cast(
    const float* __restrict__ in, __bf16* __restrict__ out, int R, int C) {
    __shared__ __bf16 t[32][33];
    const int bc = blockIdx.x * 32, br = blockIdx.y * 32;
    const int tx = threadIdx.x & 31, ty = threadIdx.x >> 5;   // 32 x 8
#pragma unroll
    for (int i = 0; i < 32; i += 8)
        t[tx][ty + i] = (__bf16)in[(size_t)(br + ty + i) * C + (bc + tx)];
    __syncthreads();
#pragma unroll
    for (int i = 0; i < 32; i += 8)
        out[(size_t)(bc + ty + i) * R + (br + tx)] = t[ty + i][tx];
}

// ------------- bf16 GEMM (TN): C(MxN) = A(MxK) * Bt(NxK)^T -----------------
// Both operands K-major.  Block 256 threads = 8 waves (2x4); wave = 64x32 tile.
// TDM double-buffers the 128x32 A and B tiles into padded LDS rows.
static __device__ __forceinline__ void mma_step(
    const __bf16 (*la)[40], const __bf16 (*lb)[40],
    int wm, int wn, int fl, int koffA, int kbaseB, v8f acc[4][2]) {
    v16bf bfr[2];
#pragma unroll
    for (int nt = 0; nt < 2; ++nt) {
        const int n = wn * 32 + nt * 16 + fl;
        v8bf lo = *(const v8bf*)&lb[n][kbaseB];
        v8bf hi = *(const v8bf*)&lb[n][kbaseB + 16];
        bfr[nt] = cat8(lo, hi);
    }
#pragma unroll
    for (int mt = 0; mt < 4; ++mt) {
        const int m = wm * 64 + mt * 16 + fl;
        v8bf lo = *(const v8bf*)&la[m][koffA];
        v8bf hi = *(const v8bf*)&la[m][16 + koffA];
        v16bf af = cat8(lo, hi);
#pragma unroll
        for (int nt = 0; nt < 2; ++nt)
            acc[mt][nt] = wmma_bf16(af, bfr[nt], acc[mt][nt]);
    }
}

__global__ __launch_bounds__(256) void gemm_bf16_tn(
    const __bf16* __restrict__ A, const __bf16* __restrict__ Bt,
    float* __restrict__ C, int M, int N, int K) {
    __shared__ __bf16 lA[2][128][40];
    __shared__ __bf16 lB[2][128][40];
    const int tid  = threadIdx.x;
    const int lane = tid & 31;
    const int wave = tid >> 5;
    const int wm = wave >> 2, wn = wave & 3;
    const int bm = blockIdx.y * 128, bn = blockIdx.x * 128;
    const int fl = lane & 15;
    const int hb = (lane >> 4) & 1;
    const int koffA  = hb * 8;
    const int kbaseB = hb * 8;
    v8f acc[4][2] = {};

#ifdef HAVE_TDM
    const int steps = K >> 5;
    if (wave == 0) {
        tdm_tile_to_lds(A  + (size_t)bm * K, (unsigned)(size_t)&lA[0][0][0], K, M, 128);
        tdm_tile_to_lds(Bt + (size_t)bn * K, (unsigned)(size_t)&lB[0][0][0], K, N, 128);
        __builtin_amdgcn_s_wait_tensorcnt(0);
    }
    __syncthreads();
    for (int it = 0; it < steps; ++it) {
        const int cur = it & 1;
        if (wave == 0 && it + 1 < steps) {
            const int k1 = (it + 1) << 5;
            tdm_tile_to_lds(A  + (size_t)bm * K + k1,
                            (unsigned)(size_t)&lA[cur ^ 1][0][0], K, M, 128);
            tdm_tile_to_lds(Bt + (size_t)bn * K + k1,
                            (unsigned)(size_t)&lB[cur ^ 1][0][0], K, N, 128);
        }
        mma_step(lA[cur], lB[cur], wm, wn, fl, koffA, kbaseB, acc);
        if (wave == 0) __builtin_amdgcn_s_wait_tensorcnt(0);
        __syncthreads();
    }
#else
    // synchronous fallback: global -> VGPR -> LDS (single buffer)
    const int ra = tid >> 1, ca = (tid & 1) * 16;
    for (int k0 = 0; k0 < K; k0 += 32) {
        const __bf16* as = A  + (size_t)(bm + ra) * K + k0 + ca;
        const __bf16* bs = Bt + (size_t)(bn + ra) * K + k0 + ca;
        v8bf a0 = *(const v8bf*)as, a1 = *(const v8bf*)(as + 8);
        v8bf b0 = *(const v8bf*)bs, b1 = *(const v8bf*)(bs + 8);
        *(v8bf*)&lA[0][ra][ca]     = a0;
        *(v8bf*)&lA[0][ra][ca + 8] = a1;
        *(v8bf*)&lB[0][ra][ca]     = b0;
        *(v8bf*)&lB[0][ra][ca + 8] = b1;
        __syncthreads();
        mma_step(lA[0], lB[0], wm, wn, fl, koffA, kbaseB, acc);
        __syncthreads();
    }
#endif
    // C layout: VGPR r -> row r (+8 for upper half-lanes), N = lane&15
#pragma unroll
    for (int mt = 0; mt < 4; ++mt)
#pragma unroll
        for (int nt = 0; nt < 2; ++nt) {
            const int col = bn + wn * 32 + nt * 16 + fl;
#pragma unroll
            for (int r = 0; r < 8; ++r) {
                const int row = bm + wm * 64 + mt * 16 + hb * 8 + r;
                C[(size_t)row * N + col] = acc[mt][nt][r];
            }
        }
}

// ------------------------------- RoPE (f32, in place) ----------------------
__global__ void rope_f32(float* __restrict__ t, int nheads,
                         const int* __restrict__ spp) {
    const int start = *spp;
    const int total = ROWS_ * nheads * (HD_ / 2);
    for (int i = blockIdx.x * blockDim.x + threadIdx.x; i < total;
         i += gridDim.x * blockDim.x) {
        const int pair = i & 63;
        const int head = (i >> 6) % nheads;
        const int row  = i / (64 * nheads);
        const int pos  = (row & (SEQ_ - 1)) + start;
        const float inv = __expf(-9.210340371976184f * (float)pair * (1.0f / 64.0f));
        const float ang = (float)pos * inv;
        const float c = __cosf(ang), s = __sinf(ang);
        float* p = t + ((size_t)row * nheads + head) * HD_ + pair * 2;
        const float tr = p[0], ti = p[1];
        p[0] = tr * c - ti * s;
        p[1] = tr * s + ti * c;
    }
}

// ----------------- rearrange K -> [b,kvh,s,d], V -> [b,kvh,d,s] (bf16) -----
__global__ void arrange_kc(const float* __restrict__ kf, __bf16* __restrict__ kc) {
    const int total = ROWS_ * NKV_ * HD_;
    for (int i = blockIdx.x * blockDim.x + threadIdx.x; i < total;
         i += gridDim.x * blockDim.x) {
        const int d = i & 127, kvh = (i >> 7) & 7, row = i >> 10;
        const int b = row >> 11, s = row & (SEQ_ - 1);
        kc[(((size_t)b * NKV_ + kvh) * SEQ_ + s) * HD_ + d] = (__bf16)kf[i];
    }
}
__global__ void arrange_vt(const float* __restrict__ vf, __bf16* __restrict__ vt) {
    const int total = ROWS_ * NKV_ * HD_;
    for (int i = blockIdx.x * blockDim.x + threadIdx.x; i < total;
         i += gridDim.x * blockDim.x) {
        const int d = i & 127, kvh = (i >> 7) & 7, row = i >> 10;
        const int b = row >> 11, s = row & (SEQ_ - 1);
        vt[(((size_t)b * NKV_ + kvh) * HD_ + d) * SEQ_ + s] = (__bf16)vf[i];
    }
}

// -------------------- flash attention (online softmax, WMMA) ---------------
// grid (32 qblocks, 32 heads, 2 batch), block 128 = 4 waves, wave = 16 q rows.
__global__ __launch_bounds__(128) void flash_attn(
    const __bf16* __restrict__ qbf, const __bf16* __restrict__ kc,
    const __bf16* __restrict__ vt, __bf16* __restrict__ outb) {
    __shared__ __bf16 pt[4][16][40];     // per-wave P transpose staging
    const int lane = threadIdx.x & 31;
    const int wave = threadIdx.x >> 5;
    const int fl = lane & 15;
    const int hb = (lane >> 4) & 1;
    const int koffA = hb * 8;
    const int kbaseB = hb * 8;
    const int qb = blockIdx.x, h = blockIdx.y, b = blockIdx.z;
    const int kvh = h >> 2;                      // N_REP = 4
    const int qrow0 = qb * 64 + wave * 16;

    const __bf16* qbase = qbf + (size_t)(b * SEQ_ + qrow0 + fl) * DIM_ + h * HD_;
    v16bf qf[4];
#pragma unroll
    for (int kt = 0; kt < 4; ++kt) {
        v8bf lo = *(const v8bf*)(qbase + kt * 32 + koffA);
        v8bf hi = *(const v8bf*)(qbase + kt * 32 + 16 + koffA);
        qf[kt] = cat8(lo, hi);
    }
    const __bf16* kcb = kc + (size_t)(b * NKV_ + kvh) * SEQ_ * HD_;
    const __bf16* vtb = vt + (size_t)(b * NKV_ + kvh) * HD_ * SEQ_;

    v8f o[8] = {};
    float mrow[8], lrow[8];
#pragma unroll
    for (int r = 0; r < 8; ++r) { mrow[r] = -3.0e38f; lrow[r] = 0.0f; }
    const float scale = 0.08838834764831845f;    // 1/sqrt(128)
    const int tend = qrow0 + 16;                 // causal: kv pos <= q pos

    for (int t0 = 0; t0 < tend; t0 += 32) {
        v8f sc[2] = {};
#pragma unroll
        for (int kt = 0; kt < 4; ++kt) {
#pragma unroll
            for (int ct = 0; ct < 2; ++ct) {
                const __bf16* ks = kcb + (size_t)(t0 + ct * 16 + fl) * HD_
                                       + kt * 32 + kbaseB;
                v8bf lo = *(const v8bf*)ks;
                v8bf hi = *(const v8bf*)(ks + 16);
                sc[ct] = wmma_bf16(qf[kt], cat8(lo, hi), sc[ct]);
            }
        }
        float p0a[8], p1a[8];
#pragma unroll
        for (int r = 0; r < 8; ++r) {
            const int qpos = qrow0 + hb * 8 + r;
            float v0 = sc[0][r] * scale;
            float v1 = sc[1][r] * scale;
            if (t0 + fl      > qpos) v0 = -1.0e30f;
            if (t0 + 16 + fl > qpos) v1 = -1.0e30f;
            float tm = fmaxf(v0, v1);
            tm = fmaxf(tm, __shfl_xor(tm, 1, 32));
            tm = fmaxf(tm, __shfl_xor(tm, 2, 32));
            tm = fmaxf(tm, __shfl_xor(tm, 4, 32));
            tm = fmaxf(tm, __shfl_xor(tm, 8, 32));
            const float mn  = fmaxf(mrow[r], tm);
            const float fct = __expf(mrow[r] - mn);
            const float p0 = __expf(v0 - mn);
            const float p1 = __expf(v1 - mn);
            float rs = p0 + p1;
            rs += __shfl_xor(rs, 1, 32);
            rs += __shfl_xor(rs, 2, 32);
            rs += __shfl_xor(rs, 4, 32);
            rs += __shfl_xor(rs, 8, 32);
            lrow[r] = lrow[r] * fct + rs;
            mrow[r] = mn;
            p0a[r] = p0; p1a[r] = p1;
#pragma unroll
            for (int nt = 0; nt < 8; ++nt) o[nt][r] *= fct;
        }
        // C layout -> A layout via per-wave LDS tile
#pragma unroll
        for (int r = 0; r < 8; ++r) {
            pt[wave][hb * 8 + r][fl]      = (__bf16)p0a[r];
            pt[wave][hb * 8 + r][16 + fl] = (__bf16)p1a[r];
        }
        asm volatile("s_wait_dscnt 0" ::: "memory");
        v8bf plo = *(const v8bf*)&pt[wave][fl][koffA];
        v8bf phi = *(const v8bf*)&pt[wave][fl][16 + koffA];
        v16bf pa = cat8(plo, phi);
#pragma unroll
        for (int nt = 0; nt < 8; ++nt) {
            const __bf16* vs = vtb + (size_t)(nt * 16 + fl) * SEQ_ + t0 + kbaseB;
            v8bf vlo = *(const v8bf*)vs;
            v8bf vhi = *(const v8bf*)(vs + 16);
            o[nt] = wmma_bf16(pa, cat8(vlo, vhi), o[nt]);
        }
    }
#pragma unroll
    for (int r = 0; r < 8; ++r) {
        const float inv = 1.0f / lrow[r];
        const size_t row = (size_t)(b * SEQ_ + qrow0 + hb * 8 + r);
#pragma unroll
        for (int nt = 0; nt < 8; ++nt)
            outb[row * DIM_ + h * HD_ + nt * 16 + fl] = (__bf16)(o[nt][r] * inv);
    }
}

// ---------------------------------------------------------------------------
extern "C" void kernel_launch(void* const* d_in, const int* in_sizes, int n_in,
                              void* d_out, int out_size, void* d_ws, size_t ws_size,
                              hipStream_t stream) {
    const float* x  = (const float*)d_in[0];
    const float* wq = (const float*)d_in[1];
    const float* wk = (const float*)d_in[2];
    const float* wv = (const float*)d_in[3];
    const float* wo = (const float*)d_in[4];
    /* d_in[5] = mask: causal triu applied analytically in flash_attn */
    const int* spp  = (const int*)d_in[6];

    char* ws = (char*)d_ws;
    size_t off = 0;
    auto alloc = [&](size_t bytes) {
        char* p = ws + off;
        off += (bytes + 255) & ~(size_t)255;
        return p;
    };
    __bf16* xh  = (__bf16*)alloc((size_t)ROWS_ * DIM_ * 2);
    __bf16* wqT = (__bf16*)alloc((size_t)DIM_ * DIM_ * 2);    // [4096][4096]
    __bf16* wkT = (__bf16*)alloc((size_t)DIM_ * 1024 * 2);    // [1024][4096]
    __bf16* wvT = (__bf16*)alloc((size_t)DIM_ * 1024 * 2);    // [1024][4096]
    __bf16* woT = (__bf16*)alloc((size_t)DIM_ * DIM_ * 2);    // [4096][4096]
    float*  qf4 = (float*)alloc((size_t)ROWS_ * DIM_ * 4);
    float*  kf4 = (float*)alloc((size_t)ROWS_ * 1024 * 4);
    float*  vf4 = (float*)alloc((size_t)ROWS_ * 1024 * 4);
    // region reuse (stream-ordered; producers of reused regions are done)
    __bf16* qbf   = xh;             // after Q/K/V GEMMs, xh is free
    __bf16* kcch  = wkT;            // after K GEMM, wkT is free
    __bf16* vtch  = wvT;            // after V GEMM, wvT is free
    __bf16* attnb = (__bf16*)qf4;   // after q cast, qf4 is free

    const int nXW = ROWS_ * DIM_;   // 16,777,216

    cast_f32_bf16<<<2048, 256, 0, stream>>>(x, xh, nXW);
    transpose_cast<<<dim3(DIM_ / 32, DIM_ / 32), 256, 0, stream>>>(wq, wqT, DIM_, DIM_);
    transpose_cast<<<dim3(1024 / 32, DIM_ / 32), 256, 0, stream>>>(wk, wkT, DIM_, 1024);
    transpose_cast<<<dim3(1024 / 32, DIM_ / 32), 256, 0, stream>>>(wv, wvT, DIM_, 1024);
    transpose_cast<<<dim3(DIM_ / 32, DIM_ / 32), 256, 0, stream>>>(wo, woT, DIM_, DIM_);

    gemm_bf16_tn<<<dim3(DIM_ / 128, ROWS_ / 128), 256, 0, stream>>>(
        xh, wqT, qf4, ROWS_, DIM_, DIM_);
    gemm_bf16_tn<<<dim3(1024 / 128, ROWS_ / 128), 256, 0, stream>>>(
        xh, wkT, kf4, ROWS_, 1024, DIM_);
    gemm_bf16_tn<<<dim3(1024 / 128, ROWS_ / 128), 256, 0, stream>>>(
        xh, wvT, vf4, ROWS_, 1024, DIM_);

    rope_f32<<<2048, 256, 0, stream>>>(qf4, NH_,  spp);
    rope_f32<<<2048, 256, 0, stream>>>(kf4, NKV_, spp);

    arrange_kc<<<2048, 256, 0, stream>>>(kf4, kcch);
    arrange_vt<<<2048, 256, 0, stream>>>(vf4, vtch);
    cast_f32_bf16<<<2048, 256, 0, stream>>>(qf4, qbf, nXW);

    flash_attn<<<dim3(SEQ_ / 64, NH_, BATCH_), 128, 0, stream>>>(
        qbf, kcch, vtch, attnb);

    gemm_bf16_tn<<<dim3(DIM_ / 128, ROWS_ / 128), 256, 0, stream>>>(
        attnb, woT, (float*)d_out, ROWS_, DIM_, DIM_);
}
